// SRWMlayer_90537910599708
// MI455X (gfx1250) — matrix operand order repaired and checked by reading, not compiled, not run on hardware
//
#include <hip/hip_runtime.h>
#include <hip/hip_bf16.h>

// Problem constants (match reference)
#define SLEN   256
#define BSZ    16
#define NH     8
#define DD     64
#define IN_DIM 512
#define NE     208   // concatenated output dim: 64(Wy)+64(Wq)+64(Wk)+16(Wb padded)

typedef __attribute__((ext_vector_type(16))) __bf16 v16bf;
typedef __attribute__((ext_vector_type(8)))  float  v8f;

static __device__ __forceinline__ float fast_rcp(float x) {
  return __builtin_amdgcn_rcpf(x);
}

// ---------------------------------------------------------------------------
// WMMA operand loaders (bf16 16x16x32, wave32 layouts per CDNA5 ISA 7.12.2)
//
// A (16x32): lane L holds row M=L%16; elements e<8  -> K = kk + 8*(L/16)+e
//                                     elements e>=8 -> K = kk + 16 + 8*(L/16)+(e-8)
// B (32x16): lane L holds col N=L%16; same K mapping as A
// C (16x16 f32): VGPR v: lanes 0-15 -> M=v, lanes 16-31 -> M=v+8; N = lane%16
// ---------------------------------------------------------------------------

static __device__ __forceinline__ v16bf ld_A_rows(const float* rows, int ldr,
                                                  int nrows, int kk, int lane) {
  v16bf a = {};
  const int m = lane & 15;
  if (m < nrows) {
    const float* base = rows + (size_t)m * ldr + kk + 8 * (lane >> 4);
    float4 f0 = ((const float4*)base)[0];
    float4 f1 = ((const float4*)base)[1];
    float4 f2 = ((const float4*)(base + 16))[0];
    float4 f3 = ((const float4*)(base + 16))[1];
    a[0]  = (__bf16)f0.x; a[1]  = (__bf16)f0.y; a[2]  = (__bf16)f0.z; a[3]  = (__bf16)f0.w;
    a[4]  = (__bf16)f1.x; a[5]  = (__bf16)f1.y; a[6]  = (__bf16)f1.z; a[7]  = (__bf16)f1.w;
    a[8]  = (__bf16)f2.x; a[9]  = (__bf16)f2.y; a[10] = (__bf16)f2.z; a[11] = (__bf16)f2.w;
    a[12] = (__bf16)f3.x; a[13] = (__bf16)f3.y; a[14] = (__bf16)f3.z; a[15] = (__bf16)f3.w;
  }
  return a;
}

// B[k][n] sourced from S[n][k] (row-major, leading dim ldb): two 8-float runs/lane (A's K order).
static __device__ __forceinline__ v16bf ld_B_cols(const float* S, int ldb,
                                                  int n0, int kk, int lane) {
  const float4* p = (const float4*)(S + (size_t)(n0 + (lane & 15)) * ldb +
                                    kk + 8 * (lane >> 4));
  v16bf b = {};
#pragma unroll
  for (int j = 0; j < 4; ++j) {
    float4 f = p[j < 2 ? j : j + 2];
    b[4 * j + 0] = (__bf16)f.x; b[4 * j + 1] = (__bf16)f.y;
    b[4 * j + 2] = (__bf16)f.z; b[4 * j + 3] = (__bf16)f.w;
  }
  return b;
}

// ---------------------------------------------------------------------------
// Stage 1: the sequential SRWM scan. One workgroup per (b,h); state in LDS.
// ---------------------------------------------------------------------------
__global__ __launch_bounds__(256) void srwm_scan(
    const float* __restrict__ hin, const float* __restrict__ Wy,
    const float* __restrict__ Wq,  const float* __restrict__ Wk,
    const float* __restrict__ wb,  float* __restrict__ ys) {
  __shared__ __align__(16) float WT[NE * 64];  // W^T: row e (output), col d (input), fp32 master
  __shared__ __align__(16) float vecs[3 * 64]; // rows: 0=q, 1=k, 2=x
  __shared__ float preA[NE];
  __shared__ float dvv[NE];
  __shared__ float beta4[4];

  const int tid  = threadIdx.x;
  const int lane = tid & 31;
  const int wv   = tid >> 5;      // 8 waves
  const int b    = blockIdx.x >> 3;
  const int hh   = blockIdx.x & 7;

  // ---- init WT from broadcast initial weights ----
  for (int idx = tid; idx < NE * 64; idx += 256) {
    const int e = idx >> 6, d = idx & 63;
    float v = 0.f;
    if (e < 64)       v = Wy[(hh * 64 + d) * 64 + e];
    else if (e < 128) v = Wq[(hh * 64 + d) * 64 + (e - 64)];
    else if (e < 192) v = Wk[(hh * 64 + d) * 64 + (e - 128)];
    else if (e < 196) v = wb[(hh * 64 + d) * 4 + (e - 192)];
    WT[idx] = v;
  }
  __syncthreads();

  for (int s = 0; s < SLEN; ++s) {
    // ---- async load x_t straight into LDS (waves 0,1; ASYNCcnt-tracked) ----
    const float* xg = hin + (size_t)(s * BSZ + b) * IN_DIM + hh * 64;
    if (tid < 64) {   // wave-uniform: waves 0 and 1 exactly
      const float* gp = xg + tid;
      unsigned loff = (unsigned)(unsigned long long)(&vecs[128 + tid]);
      asm volatile("global_load_async_to_lds_b32 %0, %1, off"
                   :: "v"(loff), "v"(gp) : "memory");
      if (s + 1 < SLEN)
        __builtin_prefetch(hin + (size_t)((s + 1) * BSZ + b) * IN_DIM + hh * 64 + tid, 0, 0);
      asm volatile("s_wait_asynccnt 0x0" ::: "memory");
    }
    __syncthreads();

    // ---- Phase A: x . [Wq | Wk | Wb]  (col blocks 4..12) ----
    {
      v16bf ax0 = ld_A_rows(vecs + 128, 64, 1, 0, lane);
      v16bf ax1 = ld_A_rows(vecs + 128, 64, 1, 32, lane);
      for (int blk = 4 + wv; blk < 13; blk += 8) {
        v8f c = {};
        v16bf b0 = ld_B_cols(WT, 64, blk * 16, 0, lane);
        c = __builtin_amdgcn_wmma_f32_16x16x32_bf16(false, ax0, false, b0, (short)0, c, false, false);
        v16bf b1 = ld_B_cols(WT, 64, blk * 16, 32, lane);
        c = __builtin_amdgcn_wmma_f32_16x16x32_bf16(false, ax1, false, b1, (short)0, c, false, false);
        if (lane < 16) preA[blk * 16 + lane] = c[0];   // M=0 row = x . W
      }
    }
    __syncthreads();

    // ---- softmax(q), softmax(k), sigmoid(beta) ----
    if (wv < 2) {                       // wave0 -> q, wave1 -> k
      const float* src = preA + 64 + wv * 64;
      float a0 = src[lane], a1 = src[lane + 32];
      float mx = fmaxf(a0, a1);
#pragma unroll
      for (int off = 16; off >= 1; off >>= 1) mx = fmaxf(mx, __shfl_xor(mx, off));
      float e0 = __expf(a0 - mx), e1 = __expf(a1 - mx);
      float sm = e0 + e1;
#pragma unroll
      for (int off = 16; off >= 1; off >>= 1) sm += __shfl_xor(sm, off);
      const float inv = fast_rcp(sm);
      vecs[wv * 64 + lane]      = e0 * inv;
      vecs[wv * 64 + lane + 32] = e1 * inv;
    } else if (wv == 2 && lane < 4) {
      beta4[lane] = fast_rcp(1.f + __expf(-preA[192 + lane]));
    }
    __syncthreads();

    // ---- Phase B: [q; k; x] . [Wy|Wq|Wk|Wb]  (col blocks 0..12) ----
    {
      v16bf aq0 = ld_A_rows(vecs, 64, 3, 0, lane);
      v16bf aq1 = ld_A_rows(vecs, 64, 3, 32, lane);
      for (int blk = wv; blk < 13; blk += 8) {
        v8f c = {};
        v16bf b0 = ld_B_cols(WT, 64, blk * 16, 0, lane);
        c = __builtin_amdgcn_wmma_f32_16x16x32_bf16(false, aq0, false, b0, (short)0, c, false, false);
        v16bf b1 = ld_B_cols(WT, 64, blk * 16, 32, lane);
        c = __builtin_amdgcn_wmma_f32_16x16x32_bf16(false, aq1, false, b1, (short)0, c, false, false);
        if (lane < 16) {
          const float bs = beta4[blk >> 2];            // 0-3:Wy 4-7:Wq 8-11:Wk 12:Wb
          dvv[blk * 16 + lane] = bs * (c[0] - c[1]);   // beta * (vq - vk)
          if (blk < 4)                                  // y = x . Wy (row M=2)
            ys[(size_t)(s * BSZ + b) * IN_DIM + hh * 64 + blk * 16 + lane] = c[2];
        }
      }
    }
    __syncthreads();

    // ---- Phase C: rank-1 update  WT[e][d] += dv[e] * k[d]  (float2 per lane) ----
    {
      const int dl = (tid & 31) * 2;   // column pair: 32 lanes cover all 64 banks, 8B each
      const float2 kd = *(const float2*)&vecs[64 + dl];
      for (int e = wv; e < NE; e += 8) {
        const float dv = dvv[e];
        float2* w = (float2*)&WT[e * 64 + dl];
        float2 val = *w;
        val.x += dv * kd.x;
        val.y += dv * kd.y;
        *w = val;
      }
    }
    __syncthreads();
  }
}

// ---------------------------------------------------------------------------
// Stage 2a: per-row LayerNorm stats of h (4096 rows of 512)
// ---------------------------------------------------------------------------
__global__ __launch_bounds__(256) void srwm_lnstats(const float* __restrict__ hin,
                                                    float* __restrict__ mu,
                                                    float* __restrict__ rs) {
  const int row  = blockIdx.x * 8 + (threadIdx.x >> 5);
  const int lane = threadIdx.x & 31;
  const float* x = hin + (size_t)row * IN_DIM;
  float s = 0.f, s2 = 0.f;
#pragma unroll
  for (int i = lane; i < IN_DIM; i += 32) { float v = x[i]; s += v; s2 += v * v; }
#pragma unroll
  for (int off = 16; off >= 1; off >>= 1) { s += __shfl_xor(s, off); s2 += __shfl_xor(s2, off); }
  const float m   = s * (1.f / IN_DIM);
  const float var = fmaxf(s2 * (1.f / IN_DIM) - m * m, 0.f);
  if (lane == 0) { mu[row] = m; rs[row] = __builtin_amdgcn_rsqf(var + 1e-5f); }
}

// ---------------------------------------------------------------------------
// Stage 2b: out = ys @ W_out^T + LayerNorm(h).  WMMA GEMM, LN fused epilogue.
// Grid: (4096/64, 512/64); block 128 (4 waves, each: 16 rows x 64 cols).
// ---------------------------------------------------------------------------
__global__ __launch_bounds__(128) void srwm_out(
    const float* __restrict__ ys,  const float* __restrict__ Wout,
    const float* __restrict__ hin, const float* __restrict__ mu,
    const float* __restrict__ rs,  const float* __restrict__ g,
    const float* __restrict__ bb,  float* __restrict__ out) {
  const int lane = threadIdx.x & 31;
  const int wv   = threadIdx.x >> 5;
  const int m0   = blockIdx.x * 64 + wv * 16;
  const int n0   = blockIdx.y * 64;

  v8f c0 = {}, c1 = {}, c2 = {}, c3 = {};
  for (int kk = 0; kk < IN_DIM; kk += 32) {
    v16bf a  = ld_A_rows(ys + (size_t)m0 * IN_DIM, IN_DIM, 16, kk, lane);
    v16bf b0 = ld_B_cols(Wout, IN_DIM, n0 +  0, kk, lane);
    c0 = __builtin_amdgcn_wmma_f32_16x16x32_bf16(false, a, false, b0, (short)0, c0, false, false);
    v16bf b1 = ld_B_cols(Wout, IN_DIM, n0 + 16, kk, lane);
    c1 = __builtin_amdgcn_wmma_f32_16x16x32_bf16(false, a, false, b1, (short)0, c1, false, false);
    v16bf b2 = ld_B_cols(Wout, IN_DIM, n0 + 32, kk, lane);
    c2 = __builtin_amdgcn_wmma_f32_16x16x32_bf16(false, a, false, b2, (short)0, c2, false, false);
    v16bf b3 = ld_B_cols(Wout, IN_DIM, n0 + 48, kk, lane);
    c3 = __builtin_amdgcn_wmma_f32_16x16x32_bf16(false, a, false, b3, (short)0, c3, false, false);
  }

#pragma unroll
  for (int v = 0; v < 8; ++v) {
    const int r = m0 + v + ((lane >> 4) << 3);   // M = v (+8 for upper half-lanes)
    const float m = mu[r], ir = rs[r];
    const float* hr = hin + (size_t)r * IN_DIM;
    float* orow = out + (size_t)r * IN_DIM;
#pragma unroll
    for (int nb = 0; nb < 4; ++nb) {
      const int o = n0 + nb * 16 + (lane & 15);
      const float acc = (nb == 0) ? c0[v] : (nb == 1) ? c1[v] : (nb == 2) ? c2[v] : c3[v];
      orow[o] = acc + (hr[o] - m) * ir * g[o] + bb[o];
    }
  }
}

// ---------------------------------------------------------------------------
extern "C" void kernel_launch(void* const* d_in, const int* in_sizes, int n_in,
                              void* d_out, int out_size, void* d_ws, size_t ws_size,
                              hipStream_t stream) {
  (void)in_sizes; (void)n_in; (void)out_size; (void)ws_size;
  const float* hin  = (const float*)d_in[0];
  const float* Wy   = (const float*)d_in[1];
  const float* Wq   = (const float*)d_in[2];
  const float* Wk   = (const float*)d_in[3];
  const float* wb   = (const float*)d_in[4];
  const float* Wout = (const float*)d_in[5];
  const float* g    = (const float*)d_in[6];
  const float* bb   = (const float*)d_in[7];
  float* out = (float*)d_out;

  float* ys = (float*)d_ws;                       // 4096*512 floats (8 MB)
  float* mu = ys + (size_t)SLEN * BSZ * IN_DIM;   // 4096 floats
  float* rs = mu + SLEN * BSZ;                    // 4096 floats

  srwm_scan<<<BSZ * NH, 256, 0, stream>>>(hin, Wy, Wq, Wk, wb, ys);
  srwm_lnstats<<<(SLEN * BSZ) / 8, 256, 0, stream>>>(hin, mu, rs);
  srwm_out<<<dim3((SLEN * BSZ) / 64, IN_DIM / 64), 128, 0, stream>>>(
      ys, Wout, hin, mu, rs, g, bb, out);
}
